// GraphEncoder_82575041233261
// MI455X (gfx1250) — compile-verified, matrix operand order checked by reading, not run on hardware
//
#include <hip/hip_runtime.h>

typedef _Float16 half_t;
typedef __attribute__((ext_vector_type(16))) _Float16 v16h;
typedef __attribute__((ext_vector_type(8)))  _Float16 v8h;
typedef __attribute__((ext_vector_type(8)))  float    v8f;

#define N0     32768
#define EDGES  262144
#define NGRAPH 32

// ---------------- device helpers ----------------
static __device__ __forceinline__ v16h ldfrag(const half_t* p){
  v8h lo = *(const v8h*)(p);
  v8h hi = *(const v8h*)(p + 16);
  v16h r;
#pragma unroll
  for (int i = 0; i < 8; ++i){ r[i] = lo[i]; r[i + 8] = hi[i]; }
  return r;
}
static __device__ __forceinline__ v8f wmma_f16(v16h a, v16h b, v8f c){
  // D = A(16x32 f16) x B(32x16 f16) + C(f32)
  return __builtin_amdgcn_wmma_f32_16x16x32_f16(false, a, false, b, (short)0, c, false, false);
}
static __device__ __forceinline__ unsigned ford(float f){
  unsigned u = __float_as_uint(f);
  return (u & 0x80000000u) ? ~u : (u | 0x80000000u);
}
static __device__ __forceinline__ float funord(unsigned e){
  return __uint_as_float((e & 0x80000000u) ? (e & 0x7fffffffu) : ~e);
}

// ---------------- utility kernels ----------------
__global__ void fill_f32_k(float* p, float v, size_t n){
  size_t i = (size_t)blockIdx.x * 256 + threadIdx.x;
  if (i < n) p[i] = v;
}
__global__ void fill_i32_k(int* p, int v, size_t n){
  size_t i = (size_t)blockIdx.x * 256 + threadIdx.x;
  if (i < n) p[i] = v;
}
__global__ void fill_enc_k(unsigned* p, float v, size_t n){
  size_t i = (size_t)blockIdx.x * 256 + threadIdx.x;
  if (i < n) p[i] = ford(v);
}
__global__ void cvt_k(const float* __restrict__ s, half_t* __restrict__ d, size_t n){
  size_t i = (size_t)blockIdx.x * 256 + threadIdx.x;
  if (i < n) d[i] = (half_t)s[i];
}
// transpose-convert W (K x N, f32) -> WT (N x K, f16)
__global__ void tcw_k(const float* __restrict__ W, half_t* __restrict__ WT, int K, int N){
  int i = blockIdx.x * 256 + threadIdx.x;
  if (i >= K * N) return;
  int k = i / N, c = i - k * N;
  WT[(size_t)c * K + k] = (half_t)W[i];
}
__global__ void copy_edges_k(const int* __restrict__ ei, const float* __restrict__ ew,
                             int* src, int* dst, float* ewc, int* mask){
  int i = blockIdx.x * 256 + threadIdx.x;
  if (i >= EDGES) return;
  src[i] = ei[i]; dst[i] = ei[EDGES + i]; ewc[i] = ew[i]; mask[i] = 1;
}

// ---------------- GEMM: Y[n,NT] = Xh[n,K] @ WT^T + bias ----------------
// WT stored as (NT x K) f16. One wave computes a 16x64 tile.
__global__ __launch_bounds__(32)
void gemm_wmma_k(const half_t* __restrict__ Xh, const half_t* __restrict__ WT,
                 const float* __restrict__ bias, float* __restrict__ Y,
                 half_t* __restrict__ Yh, int K, int NT, int relu){
  int lane = threadIdx.x;
  int m0 = blockIdx.x * 16;
  int n0 = blockIdx.y * 64;
  int r = lane & 15, hf = lane >> 4;
  const half_t* arow = Xh + (size_t)(m0 + r) * K;
  const half_t* brow = WT + (size_t)(n0 + r) * K;
  v8f acc[4] = {};
  for (int k0 = 0; k0 < K; k0 += 32){
    v16h a = ldfrag(arow + k0 + hf * 8);
#pragma unroll
    for (int t = 0; t < 4; ++t){
      v16h b = ldfrag(brow + (size_t)t * 16 * K + k0 + hf * 8);
      acc[t] = wmma_f16(a, b, acc[t]);
    }
  }
#pragma unroll
  for (int t = 0; t < 4; ++t){
    int col = n0 + t * 16 + r;
    float bv = bias ? bias[col] : 0.f;
#pragma unroll
    for (int v = 0; v < 8; ++v){
      int row = m0 + v + 8 * hf;
      float val = acc[t][v] + bv;
      if (relu) val = fmaxf(val, 0.f);
      size_t ix = (size_t)row * NT + col;
      if (Y)  Y[ix]  = val;
      if (Yh) Yh[ix] = (half_t)val;
    }
  }
}

// ---------------- degree / norm branch ----------------
__global__ void edge_deg_k(const int* __restrict__ src, const int* __restrict__ mask, float* deg){
  int i = blockIdx.x * 256 + threadIdx.x;
  if (i >= EDGES) return;
  if (mask[i]) atomicAdd(&deg[src[i]], 1.0f);
}
__global__ void dis_k(const float* __restrict__ deg, float* dis, int n){
  int i = blockIdx.x * 256 + threadIdx.x;
  if (i >= n) return;
  float d = deg[i];
  dis[i] = (d > 0.f) ? rsqrtf(d) : 0.f;
}
__global__ void edge_xw_k(const float* __restrict__ xl, const float* __restrict__ dis,
                          const float* __restrict__ ewc, const int* __restrict__ src,
                          const int* __restrict__ dst, float* __restrict__ xw){
  int e = blockIdx.x;
  int c = threadIdx.x;           // 128 threads
  int s = src[e], d = dst[e];
  float nrm = dis[s] * ewc[e] * dis[d];
  if (nrm != 0.f)
    atomicAdd(&xw[(size_t)d * 128 + c], nrm * xl[(size_t)s * 128 + c]);
}

// ---------------- attention pass 1: alpha ----------------
// One wave handles 16 edges. eemb recomputed via WMMA (16 edges x 16 ch tiles).
__global__ __launch_bounds__(32)
void edge_alpha_k(const half_t* __restrict__ ea16, const half_t* __restrict__ eT,
                  const float* __restrict__ eb,
                  const half_t* __restrict__ q16, const half_t* __restrict__ k16,
                  const int* __restrict__ src, const int* __restrict__ dst,
                  const int* __restrict__ mask, float* __restrict__ alpha){
  int e0 = blockIdx.x * 16;
  int lane = threadIdx.x;
  __shared__ int ss[16], ds[16], ms[16];
  if (lane < 16){ ss[lane] = src[e0 + lane]; ds[lane] = dst[e0 + lane]; ms[lane] = mask[e0 + lane]; }
  __syncthreads();
  int r = lane & 15, hf = lane >> 4;
  const half_t* arow = ea16 + (size_t)(e0 + r) * 64;
  v16h af0 = ldfrag(arow + 0  + hf * 8);
  v16h af1 = ldfrag(arow + 32 + hf * 8);
  const float scale = 0.08838834764831845f;   // 1/sqrt(128)
  for (int h = 0; h < 4; ++h){
    float accv[8] = {0,0,0,0,0,0,0,0};
    for (int jj = 0; jj < 8; ++jj){
      int chunk = h * 8 + jj;
      int cc = chunk * 16 + r;
      const half_t* bcol = eT + (size_t)cc * 64;
      v8f acc = {};
      acc = wmma_f16(af0, ldfrag(bcol + 0  + hf * 8), acc);
      acc = wmma_f16(af1, ldfrag(bcol + 32 + hf * 8), acc);
      float ebv = eb[cc];
#pragma unroll
      for (int v = 0; v < 8; ++v){
        int M = v + 8 * hf;
        float ke = (float)k16[(size_t)ss[M] * 512 + cc] + acc[v] + ebv;
        float qv = (float)q16[(size_t)ds[M] * 512 + cc];
        accv[v] += qv * ke;
      }
    }
#pragma unroll
    for (int m = 1; m < 16; m <<= 1){
#pragma unroll
      for (int v = 0; v < 8; ++v) accv[v] += __shfl_xor(accv[v], m, 32);
    }
    if (r == 0){
#pragma unroll
      for (int v = 0; v < 8; ++v){
        int M = v + 8 * hf;
        alpha[(size_t)(e0 + M) * 4 + h] = ms[M] ? accv[v] * scale : -1e9f;
      }
    }
  }
}

// ---------------- segment max / softmax ----------------
__global__ void amax_k(const float* __restrict__ alpha, const int* __restrict__ dst,
                       unsigned* __restrict__ amax){
  int i = blockIdx.x * 256 + threadIdx.x;
  if (i >= EDGES * 4) return;
  int e = i >> 2, h = i & 3;
  atomicMax(&amax[(size_t)dst[e] * 4 + h], ford(alpha[i]));
}
__global__ void expa_k(float* __restrict__ alpha, const int* __restrict__ dst,
                       const int* __restrict__ mask, const unsigned* __restrict__ amax,
                       float* __restrict__ den){
  int i = blockIdx.x * 256 + threadIdx.x;
  if (i >= EDGES * 4) return;
  int e = i >> 2, h = i & 3;
  float ex = 0.f;
  if (mask[e]){
    float am = funord(amax[(size_t)dst[e] * 4 + h]);
    ex = expf(alpha[i] - am);
  }
  alpha[i] = ex;
  atomicAdd(&den[(size_t)dst[e] * 4 + h], ex);
}

// ---------------- attention pass 2: aggregate ----------------
__global__ __launch_bounds__(32)
void edge_aggr_k(const half_t* __restrict__ ea16, const half_t* __restrict__ eT,
                 const float* __restrict__ eb, const half_t* __restrict__ v16,
                 const int* __restrict__ src, const int* __restrict__ dst,
                 const float* __restrict__ expa, const float* __restrict__ den,
                 float* __restrict__ attn){
  int e0 = blockIdx.x * 16;
  int lane = threadIdx.x;
  __shared__ int ss[16], ds[16];
  if (lane < 16){ ss[lane] = src[e0 + lane]; ds[lane] = dst[e0 + lane]; }
  __syncthreads();
  int r = lane & 15, hf = lane >> 4;
  const half_t* arow = ea16 + (size_t)(e0 + r) * 64;
  v16h af0 = ldfrag(arow + 0  + hf * 8);
  v16h af1 = ldfrag(arow + 32 + hf * 8);
  for (int chunk = 0; chunk < 32; ++chunk){
    int h = chunk >> 3;
    int cc = chunk * 16 + r;
    const half_t* bcol = eT + (size_t)cc * 64;
    v8f acc = {};
    acc = wmma_f16(af0, ldfrag(bcol + 0  + hf * 8), acc);
    acc = wmma_f16(af1, ldfrag(bcol + 32 + hf * 8), acc);
    float ebv = eb[cc];
#pragma unroll
    for (int v = 0; v < 8; ++v){
      int M = v + 8 * hf;
      float a = expa[(size_t)(e0 + M) * 4 + h];
      if (a != 0.f){
        float ai = a / den[(size_t)ds[M] * 4 + h];
        float ve = (float)v16[(size_t)ss[M] * 512 + cc] + acc[v] + ebv;
        atomicAdd(&attn[(size_t)ds[M] * 512 + cc], ai * ve);
      }
    }
  }
}

// ---------------- finalize conv output ----------------
__global__ void finalize_k(float* __restrict__ attn, const float* __restrict__ sbuf,
                           const float* __restrict__ xw, half_t* __restrict__ o16, size_t total){
  size_t i = (size_t)blockIdx.x * 256 + threadIdx.x;
  if (i >= total) return;
  int c = (int)(i & 511);
  size_t node = i >> 9;
  float v = attn[i] + sbuf[i] + xw[node * 128 + (c & 127)];
  attn[i] = v;
  o16[i] = (half_t)v;
}

// ---------------- batch norm (per-column) ----------------
__global__ void bn_k(float* __restrict__ h, half_t* __restrict__ h16,
                     const float* __restrict__ g_, const float* __restrict__ b_, int n){
  int c = blockIdx.x;         // 128 blocks, one column each
  int t = threadIdx.x;        // 256 threads
  float sm = 0.f, sq = 0.f;
  for (int r = t; r < n; r += 256){
    float v = h[(size_t)r * 128 + c]; sm += v; sq += v * v;
  }
  __shared__ float s1[256], s2[256];
  s1[t] = sm; s2[t] = sq; __syncthreads();
  for (int st = 128; st > 0; st >>= 1){
    if (t < st){ s1[t] += s1[t + st]; s2[t] += s2[t + st]; }
    __syncthreads();
  }
  float mu  = s1[0] / n;
  float var = s2[0] / n - mu * mu;
  float inv = rsqrtf(var + 1e-5f) * g_[c];
  float bb  = b_[c];
  for (int r = t; r < n; r += 256){
    size_t ix = (size_t)r * 128 + c;
    float v = (h[ix] - mu) * inv + bb;
    h[ix] = v; h16[ix] = (half_t)v;
  }
}

// ---------------- pooling ----------------
__global__ void score_k(const float* __restrict__ h, const float* __restrict__ w,
                        float* __restrict__ score, int n){
  int i = blockIdx.x * 256 + threadIdx.x;
  if (i >= n) return;
  float nw = 0.f, dp = 0.f;
  for (int c = 0; c < 128; ++c){
    float wv = w[c]; nw += wv * wv;
    dp += h[(size_t)i * 128 + c] * wv;
  }
  score[i] = tanhf(dp / (sqrtf(nw) + 1e-16f));
}
__global__ void topk_k(const float* __restrict__ score, float* __restrict__ selval,
                       int* __restrict__ selorig, int* __restrict__ new_id,
                       int n_per, int kk){
  __shared__ float sc[1024];
  __shared__ int   si[1024];
  int g = blockIdx.x, t = threadIdx.x;
  for (int i = t; i < 1024; i += 256){
    sc[i] = (i < n_per) ? score[(size_t)g * n_per + i] : -1e30f;
    si[i] = i;
  }
  __syncthreads();
  for (int k = 2; k <= 1024; k <<= 1){
    for (int j = k >> 1; j > 0; j >>= 1){
      for (int i = t; i < 1024; i += 256){
        int ixj = i ^ j;
        if (ixj > i){
          bool desc = ((i & k) == 0);
          float a = sc[i], b = sc[ixj];
          bool sw = desc ? (a < b) : (a > b);
          if (sw){
            sc[i] = b; sc[ixj] = a;
            int tm = si[i]; si[i] = si[ixj]; si[ixj] = tm;
          }
        }
      }
      __syncthreads();
    }
  }
  for (int i = t; i < kk; i += 256){
    int orig = g * n_per + si[i];
    int nr = g * kk + i;
    selval[nr] = sc[i];
    selorig[nr] = orig;
    new_id[orig] = nr;
  }
}
__global__ void gather_k(const float* __restrict__ hsrc, const int* __restrict__ selorig,
                         const float* __restrict__ selval, float* __restrict__ hdst,
                         half_t* __restrict__ h16, int newn){
  int i = blockIdx.x * 256 + threadIdx.x;
  if (i >= newn * 128) return;
  int rr = i >> 7, c = i & 127;
  float v = hsrc[(size_t)selorig[rr] * 128 + c] * selval[rr];
  hdst[i] = v; h16[i] = (half_t)v;
}
__global__ void remap_k(int* src, int* dst, float* ewc, int* mask, const int* __restrict__ new_id){
  int i = blockIdx.x * 256 + threadIdx.x;
  if (i >= EDGES) return;
  int nr = new_id[src[i]], nc = new_id[dst[i]];
  int keep = (nr >= 0) && (nc >= 0) && mask[i];
  src[i] = keep ? nr : 0;
  dst[i] = keep ? nc : 0;
  ewc[i] = keep ? ewc[i] : 0.f;
  mask[i] = keep;
}
__global__ void reps_k(const float* __restrict__ h, int kk, float* __restrict__ out, int add){
  int g = blockIdx.x, c = threadIdx.x;   // 128 threads
  float mx = -1e30f, sm = 0.f;
  for (int r = 0; r < kk; ++r){
    float v = h[(size_t)(g * kk + r) * 128 + c];
    mx = fmaxf(mx, v); sm += v;
  }
  float me = sm / kk;
  if (add){ out[g * 256 + c] += mx; out[g * 256 + 128 + c] += me; }
  else    { out[g * 256 + c]  = mx; out[g * 256 + 128 + c]  = me; }
}

// ---------------- host ----------------
extern "C" void kernel_launch(void* const* d_in, const int* in_sizes, int n_in,
                              void* d_out, int out_size, void* d_ws, size_t ws_size,
                              hipStream_t stream) {
  (void)in_sizes; (void)n_in; (void)out_size; (void)ws_size;
  const float* x_in  = (const float*)d_in[0];
  const float* ea_in = (const float*)d_in[1];
  const int*   ei_in = (const int*)  d_in[2];
  const float* ew_in = (const float*)d_in[3];
  const float* lin0w = (const float*)d_in[5];
  const float* lin0b = (const float*)d_in[6];
  const float* linw  = (const float*)d_in[7];
  const float* linb  = (const float*)d_in[8];
  const float* qw = (const float*)d_in[9];
  const float* qb = (const float*)d_in[10];
  const float* kw = (const float*)d_in[11];
  const float* kb = (const float*)d_in[12];
  const float* vw = (const float*)d_in[13];
  const float* vb = (const float*)d_in[14];
  const float* eww = (const float*)d_in[15];
  const float* ewb = (const float*)d_in[16];
  const float* sw = (const float*)d_in[17];
  const float* sb = (const float*)d_in[18];
  const float* trw = (const float*)d_in[19];
  const float* trb = (const float*)d_in[20];
  const float* bng = (const float*)d_in[21];
  const float* bnb = (const float*)d_in[22];
  const float* poolw = (const float*)d_in[23];
  float* outp = (float*)d_out;

  char* base = (char*)d_ws;
  size_t cur = 0;
  auto carve = [&](size_t bytes)->char*{
    cur = (cur + 255) & ~(size_t)255;
    char* p = base + cur; cur += bytes; return p;
  };
  half_t* x16   = (half_t*)carve((size_t)N0 * 256 * 2);
  half_t* ea16  = (half_t*)carve((size_t)EDGES * 64 * 2);
  half_t* linT  = (half_t*)carve(256 * 128 * 2);
  half_t* qT    = (half_t*)carve(512 * 128 * 2);
  half_t* kT    = (half_t*)carve(512 * 128 * 2);
  half_t* vT    = (half_t*)carve(512 * 128 * 2);
  half_t* sT    = (half_t*)carve(512 * 128 * 2);
  half_t* eT    = (half_t*)carve(512 * 64 * 2);
  half_t* trT   = (half_t*)carve(128 * 512 * 2);
  float*  xl    = (float*) carve((size_t)N0 * 128 * 4);
  half_t* xl16  = (half_t*)carve((size_t)N0 * 128 * 2);
  float*  hA    = (float*) carve((size_t)N0 * 128 * 4);
  float*  hB    = (float*) carve((size_t)N0 * 128 * 4);
  half_t* h16   = (half_t*)carve((size_t)N0 * 128 * 2);
  half_t* q16b  = (half_t*)carve((size_t)N0 * 512 * 2);
  half_t* k16b  = (half_t*)carve((size_t)N0 * 512 * 2);
  half_t* v16b  = (half_t*)carve((size_t)N0 * 512 * 2);
  float*  sbuf  = (float*) carve((size_t)N0 * 512 * 4);
  float*  attn  = (float*) carve((size_t)N0 * 512 * 4);
  half_t* o16   = (half_t*)carve((size_t)N0 * 512 * 2);
  float*  alpha = (float*) carve((size_t)EDGES * 4 * 4);
  unsigned* amx = (unsigned*)carve((size_t)N0 * 4 * 4);
  float*  den   = (float*) carve((size_t)N0 * 4 * 4);
  float*  deg   = (float*) carve((size_t)N0 * 4);
  float*  dis   = (float*) carve((size_t)N0 * 4);
  float*  xw    = (float*) carve((size_t)N0 * 128 * 4);
  int*    srcA  = (int*)   carve((size_t)EDGES * 4);
  int*    dstA  = (int*)   carve((size_t)EDGES * 4);
  int*    maskA = (int*)   carve((size_t)EDGES * 4);
  float*  ewcA  = (float*) carve((size_t)EDGES * 4);
  float*  score = (float*) carve((size_t)N0 * 4);
  float*  selval= (float*) carve((size_t)N0 * 4);
  int*    selorig=(int*)   carve((size_t)N0 * 4);
  int*    new_id= (int*)   carve((size_t)N0 * 4);

  auto gblk = [](size_t n)->int{ return (int)((n + 255) / 256); };

  cvt_k<<<gblk((size_t)N0 * 256), 256, 0, stream>>>(x_in, x16, (size_t)N0 * 256);
  cvt_k<<<gblk((size_t)EDGES * 64), 256, 0, stream>>>(ea_in, ea16, (size_t)EDGES * 64);
  copy_edges_k<<<gblk(EDGES), 256, 0, stream>>>(ei_in, ew_in, srcA, dstA, ewcA, maskA);

  auto conv = [&](int l, const half_t* Xh, int Kin, const float* lw, const float* lb, int n){
    tcw_k<<<gblk((size_t)Kin * 128), 256, 0, stream>>>(lw, linT, Kin, 128);
    tcw_k<<<gblk(128 * 512), 256, 0, stream>>>(qw + (size_t)l * 128 * 512, qT, 128, 512);
    tcw_k<<<gblk(128 * 512), 256, 0, stream>>>(kw + (size_t)l * 128 * 512, kT, 128, 512);
    tcw_k<<<gblk(128 * 512), 256, 0, stream>>>(vw + (size_t)l * 128 * 512, vT, 128, 512);
    tcw_k<<<gblk(128 * 512), 256, 0, stream>>>(sw + (size_t)l * 128 * 512, sT, 128, 512);
    tcw_k<<<gblk(64 * 512), 256, 0, stream>>>(eww + (size_t)l * 64 * 512, eT, 64, 512);
    tcw_k<<<gblk(512 * 128), 256, 0, stream>>>(trw + (size_t)l * 512 * 128, trT, 512, 128);

    gemm_wmma_k<<<dim3(n / 16, 2), 32, 0, stream>>>(Xh, linT, lb, xl, xl16, Kin, 128, 0);
    gemm_wmma_k<<<dim3(n / 16, 8), 32, 0, stream>>>(xl16, qT, qb + l * 512, nullptr, q16b, 128, 512, 0);
    gemm_wmma_k<<<dim3(n / 16, 8), 32, 0, stream>>>(xl16, kT, kb + l * 512, nullptr, k16b, 128, 512, 0);
    gemm_wmma_k<<<dim3(n / 16, 8), 32, 0, stream>>>(xl16, vT, vb + l * 512, nullptr, v16b, 128, 512, 0);
    gemm_wmma_k<<<dim3(n / 16, 8), 32, 0, stream>>>(xl16, sT, sb + l * 512, sbuf, nullptr, 128, 512, 0);

    fill_f32_k<<<gblk(n), 256, 0, stream>>>(deg, 0.f, (size_t)n);
    edge_deg_k<<<gblk(EDGES), 256, 0, stream>>>(srcA, maskA, deg);
    dis_k<<<gblk(n), 256, 0, stream>>>(deg, dis, n);
    fill_f32_k<<<gblk((size_t)n * 128), 256, 0, stream>>>(xw, 0.f, (size_t)n * 128);
    edge_xw_k<<<EDGES, 128, 0, stream>>>(xl, dis, ewcA, srcA, dstA, xw);

    fill_enc_k<<<gblk((size_t)n * 4), 256, 0, stream>>>(amx, -1e30f, (size_t)n * 4);
    edge_alpha_k<<<EDGES / 16, 32, 0, stream>>>(ea16, eT, ewb + l * 512, q16b, k16b, srcA, dstA, maskA, alpha);
    amax_k<<<gblk((size_t)EDGES * 4), 256, 0, stream>>>(alpha, dstA, amx);
    fill_f32_k<<<gblk((size_t)n * 4), 256, 0, stream>>>(den, 1e-16f, (size_t)n * 4);
    expa_k<<<gblk((size_t)EDGES * 4), 256, 0, stream>>>(alpha, dstA, maskA, amx, den);
    fill_f32_k<<<gblk((size_t)n * 512), 256, 0, stream>>>(attn, 0.f, (size_t)n * 512);
    edge_aggr_k<<<EDGES / 16, 32, 0, stream>>>(ea16, eT, ewb + l * 512, v16b, srcA, dstA, alpha, den, attn);

    finalize_k<<<gblk((size_t)n * 512), 256, 0, stream>>>(attn, sbuf, xw, o16, (size_t)n * 512);
    gemm_wmma_k<<<dim3(n / 16, 2), 32, 0, stream>>>(o16, trT, trb + l * 128, hA, nullptr, 512, 128, 1);
    bn_k<<<128, 256, 0, stream>>>(hA, h16, bng + l * 128, bnb + l * 128, n);
  };

  auto pool = [&](int pi, int n, int n_per, int add){
    int kk = n_per / 2;
    int newn = n / 2;
    score_k<<<gblk(n), 256, 0, stream>>>(hA, poolw + pi * 128, score, n);
    fill_i32_k<<<gblk(n), 256, 0, stream>>>(new_id, -1, (size_t)n);
    topk_k<<<NGRAPH, 256, 0, stream>>>(score, selval, selorig, new_id, n_per, kk);
    gather_k<<<gblk((size_t)newn * 128), 256, 0, stream>>>(hA, selorig, selval, hB, h16, newn);
    remap_k<<<gblk(EDGES), 256, 0, stream>>>(srcA, dstA, ewcA, maskA, new_id);
    reps_k<<<NGRAPH, 128, 0, stream>>>(hB, kk, outp, add);
  };

  int n = N0, n_per = 1024;
  conv(0, x16, 256, lin0w, lin0b, n);
  conv(1, h16, 128, linw + 0 * 128 * 128, linb + 0 * 128, n);
  pool(0, n, n_per, 0);
  n = N0 / 2; n_per = 512;
  conv(2, h16, 128, linw + 1 * 128 * 128, linb + 1 * 128, n);
  conv(3, h16, 128, linw + 2 * 128 * 128, linb + 2 * 128, n);
  pool(1, n, n_per, 1);
}
